// CausalScaledDotAttention_14224931684505
// MI455X (gfx1250) — matrix operand order repaired and checked by reading, not compiled, not run on hardware
//
#include <hip/hip_runtime.h>
#include <hip/hip_bf16.h>

// ---------------------------------------------------------------------------
// CausalScaledDotAttention for MI455X (gfx1250, wave32, WMMA)
//   q = Q @ Wq^T + bq ; k = K @ Wk^T + bk ; v = V @ Wv^T + bv   (bf16 staged)
//   scoresT[b][t][s] = scale * <q_t, k_s> + causal(-1e7 if s>t)  (f32)
//   attnT = softmax_rows(scoresT)  -> bf16
//   attn[b][s][t] = transpose(attnT)                             (f32 out)
//   context[b][t][h] = attnT @ v                                 (f32 out)
// All matrix math through v_wmma_f32_16x16x32_bf16 (f32 accumulate).
// GEMM: 128x64 block tile, 32x32 wave tiles (4 WMMA / wave / K-step),
// register-staged global loads + double-buffered LDS (1 barrier / K-step).
// ---------------------------------------------------------------------------

typedef __attribute__((ext_vector_type(16))) __bf16 v16bf;
typedef __attribute__((ext_vector_type(2)))  __bf16 v2bf;
typedef __attribute__((ext_vector_type(2)))  float  v2f;
typedef __attribute__((ext_vector_type(8)))  float  v8f;

#define HID    1024
#define SEQ    2048
#define NBATCH 4

// native conversions: f32->bf16 fptrunc lowers to v_cvt_pk_bf16_f32 (RNE)
__device__ __forceinline__ unsigned short f2bf(float f) {
  return __builtin_bit_cast(unsigned short, (__bf16)f);
}
__device__ __forceinline__ float bf2f(unsigned short u) {
  return __uint_as_float(((unsigned int)u) << 16);
}

// raw global loads (no conversion at load time -> pure global_load_b64/b32)
__device__ __forceinline__ uint2 rawload(const float* p) {
  return *(const uint2*)p;
}
__device__ __forceinline__ unsigned int rawload(const unsigned short* p) {
  return *(const unsigned int*)p;
}
// convert raw pair -> packed bf16 dword at LDS-store time:
// <2 x f32> -> <2 x bf16> vector trunc selects a single v_cvt_pk_bf16_f32
// whose destination is already the packed dword (no repacking shuffles).
__device__ __forceinline__ unsigned int topack(uint2 r) {
  v2f f;
  f.x = __uint_as_float(r.x);
  f.y = __uint_as_float(r.y);
  return __builtin_bit_cast(unsigned int, __builtin_convertvector(f, v2bf));
}
__device__ __forceinline__ unsigned int topack(unsigned int r) { return r; }

template <typename T> struct RawOf;
template <> struct RawOf<float>          { using type = uint2; };
template <> struct RawOf<unsigned short> { using type = unsigned int; };

union Frag16 { uint4 q[2]; v16bf v; };

// Generic tiled GEMM:  C[m][n] = sum_k A[m][k] * BT[n][k]  (+bias, +causal)
template<typename SrcT, bool BIAS, bool CAUSAL, bool OUT_BF16, bool OUT_TRANS>
__global__ __launch_bounds__(256)
void gemm_bt(const SrcT* __restrict__ A, const SrcT* __restrict__ BT,
             const float* __restrict__ bias, void* __restrict__ Cout,
             int K, int lda, int ldb, int ldc, float scale,
             long strideA, long strideB, long strideC) {
  using RawT = typename RawOf<SrcT>::type;
  // Double-buffered LDS tiles, rows padded to 40 bf16 (80 B):
  //  - rows stay 16B-aligned -> fragment reads are ds_load_b128
  //  - 20-bank row rotation -> conflict-free across 16-lane groups
  __shared__ uint4 sAq[2][128 * 5];   // 128 x 32 bf16 per buffer
  __shared__ uint4 sBq[2][64 * 5];    //  64 x 32 bf16 per buffer

  const int tid   = threadIdx.x;
  const int wave  = tid >> 5;
  const int lane  = tid & 31;
  const int laneN = lane & 15;
  const int half  = lane >> 4;
  const int wr    = wave >> 1;         // 4 wave-rows  (32 M each)
  const int wc    = wave & 1;          // 2 wave-cols  (32 N each)
  const int mBase = blockIdx.y * 128;
  const int nBase = blockIdx.x * 64;

  const SrcT* Ab = A  + (size_t)blockIdx.z * strideA;
  const SrcT* Bb = BT + (size_t)blockIdx.z * strideB;

  // staging pattern: thread owns K-pair column p, rows r0 + 16*i
  const int r0 = tid >> 4;             // 0..15
  const int p  = tid & 15;             // K-pair index
  const SrcT* aPtr = Ab + (size_t)(mBase + r0) * lda + 2 * p;
  const SrcT* bPtr = Bb + (size_t)(nBase + r0) * ldb + 2 * p;

  v8f acc00 = {}, acc01 = {}, acc10 = {}, acc11 = {};

  RawT ra[8];   // A tile: 128 rows -> 8 rows per thread
  RawT rb[4];   // B tile:  64 rows -> 4 rows per thread
  #pragma unroll
  for (int i = 0; i < 8; ++i) ra[i] = rawload(aPtr + (size_t)(16 * i) * lda);
  #pragma unroll
  for (int i = 0; i < 4; ++i) rb[i] = rawload(bPtr + (size_t)(16 * i) * ldb);

  int buf = 0;
  for (int k0 = 0; k0 < K; k0 += 32) {
    // commit staged registers -> LDS buffer `buf` (bf16 convert here)
    unsigned int* sa = (unsigned int*)sAq[buf];
    unsigned int* sb = (unsigned int*)sBq[buf];
    #pragma unroll
    for (int i = 0; i < 8; ++i) sa[(r0 + 16 * i) * 20 + p] = topack(ra[i]);
    #pragma unroll
    for (int i = 0; i < 4; ++i) sb[(r0 + 16 * i) * 20 + p] = topack(rb[i]);
    __syncthreads();

    // issue next tile's global loads while WMMAs consume this one
    if (k0 + 32 < K) {
      const int kn = k0 + 32;
      #pragma unroll
      for (int i = 0; i < 8; ++i)
        ra[i] = rawload(aPtr + (size_t)(16 * i) * lda + kn);
      #pragma unroll
      for (int i = 0; i < 4; ++i)
        rb[i] = rawload(bPtr + (size_t)(16 * i) * ldb + kn);
      if (k0 + 64 < K)                        // global_prefetch_b8 next-next
        __builtin_prefetch(aPtr + (size_t)k0 + 64, 0, 0);
    }

    const uint4* a4 = sAq[buf];
    const uint4* b4 = sBq[buf];
    // A fragments (16x32 bf16): lane m<16 holds K{0..7,16..23}; m+16 holds
    // K{8..15,24..31}; each quarter is one aligned 16B ds_load_b128.
    Frag16 af0, af1, bf0, bf1;
    {
      int ar0 = wr * 32 + laneN;
      int ar1 = ar0 + 16;
      af0.q[0] = a4[ar0 * 5 + half];
      af0.q[1] = a4[ar0 * 5 + 2 + half];
      af1.q[0] = a4[ar1 * 5 + half];
      af1.q[1] = a4[ar1 * 5 + 2 + half];
    }
    // B fragments (32x16 bf16): lane n<16 holds K{0..7,16..23}, n+16 holds K{8..15,24..31}.
    {
      int br0 = wc * 32 + laneN;
      int br1 = br0 + 16;
      bf0.q[0] = b4[br0 * 5 + half];
      bf0.q[1] = b4[br0 * 5 + 2 + half];
      bf1.q[0] = b4[br1 * 5 + half];
      bf1.q[1] = b4[br1 * 5 + 2 + half];
    }
    acc00 = __builtin_amdgcn_wmma_f32_16x16x32_bf16(
        false, af0.v, false, bf0.v, (short)0, acc00, false, false);
    acc01 = __builtin_amdgcn_wmma_f32_16x16x32_bf16(
        false, af0.v, false, bf1.v, (short)0, acc01, false, false);
    acc10 = __builtin_amdgcn_wmma_f32_16x16x32_bf16(
        false, af1.v, false, bf0.v, (short)0, acc10, false, false);
    acc11 = __builtin_amdgcn_wmma_f32_16x16x32_bf16(
        false, af1.v, false, bf1.v, (short)0, acc11, false, false);
    buf ^= 1;
  }

  // Epilogue. C layout: VGPR j, lanes 0-15 -> M=j, lanes 16-31 -> M=8+j.
  #pragma unroll
  for (int mi = 0; mi < 2; ++mi) {
    #pragma unroll
    for (int ni = 0; ni < 2; ++ni) {
      v8f acc = mi ? (ni ? acc11 : acc10) : (ni ? acc01 : acc00);
      const int nGlob = nBase + wc * 32 + ni * 16 + laneN;
      float bval = 0.0f;
      if (BIAS) bval = bias[nGlob];
      #pragma unroll
      for (int j = 0; j < 8; ++j) {
        const int mGlob = mBase + wr * 32 + mi * 16 + half * 8 + j;
        float val = acc[j];
        if (BIAS) val += bval;
        if (CAUSAL) {                  // scoresT[t=m][s=n]: mask s > t
          val *= scale;
          if (nGlob > mGlob) val += -1.0e7f;
        }
        if (OUT_TRANS) {               // v-projection: write vT[b][h][s] bf16
          int bb = mGlob >> 11;        // / SEQ
          int ss = mGlob & (SEQ - 1);
          ((unsigned short*)Cout)[((size_t)(bb * HID + nGlob)) * SEQ + ss] =
              f2bf(val);
        } else if (OUT_BF16) {
          unsigned short* C =
              (unsigned short*)Cout + (size_t)blockIdx.z * strideC;
          C[(size_t)mGlob * ldc + nGlob] = f2bf(val);
        } else {
          float* C = (float*)Cout + (size_t)blockIdx.z * strideC;
          C[(size_t)mGlob * ldc + nGlob] = val;
        }
      }
    }
  }
}

// Row softmax over scoresT[b][t][:]: one wave32 per row, 8 rows per block.
// Causal mask already baked into scores (-1e7 -> exp underflows to exactly 0,
// matching the reference). Writes normalized bf16 attnT.
__global__ __launch_bounds__(256)
void softmax_rows(float* __restrict__ scoresT, unsigned short* __restrict__ attnT,
                  int S) {
  const int wave = threadIdx.x >> 5;
  const int lane = threadIdx.x & 31;
  const size_t row = (size_t)blockIdx.x * 8 + wave;
  float* srow = scoresT + row * S;

  float m = -3.4e38f;
  for (int s = lane; s < S; s += 32) m = fmaxf(m, srow[s]);
  #pragma unroll
  for (int off = 16; off; off >>= 1) m = fmaxf(m, __shfl_xor(m, off, 32));

  float sum = 0.0f;
  for (int s = lane; s < S; s += 32) {
    float e = __expf(srow[s] - m);
    srow[s] = e;                        // in-place (region rewritten later)
    sum += e;
  }
  #pragma unroll
  for (int off = 16; off; off >>= 1) sum += __shfl_xor(sum, off, 32);

  const float r = 1.0f / sum;
  unsigned short* arow = attnT + row * S;
  for (int s = lane; s < S; s += 32) arow[s] = f2bf(srow[s] * r);
}

// LDS-tiled transpose: attnT[b][t][s] (bf16) -> attn[b][s][t] (f32 output).
__global__ __launch_bounds__(256)
void transpose_attn(const unsigned short* __restrict__ attnT,
                    float* __restrict__ attn, int S) {
  __shared__ float tile[32][33];
  const size_t bOff = (size_t)blockIdx.z * S * S;
  const int t0 = blockIdx.y * 32;
  const int s0 = blockIdx.x * 32;
  for (int i = threadIdx.y; i < 32; i += 8)
    tile[i][threadIdx.x] =
        bf2f(attnT[bOff + (size_t)(t0 + i) * S + s0 + threadIdx.x]);
  __syncthreads();
  for (int i = threadIdx.y; i < 32; i += 8)
    attn[bOff + (size_t)(s0 + i) * S + t0 + threadIdx.x] = tile[threadIdx.x][i];
}

extern "C" void kernel_launch(void* const* d_in, const int* in_sizes, int n_in,
                              void* d_out, int out_size, void* d_ws, size_t ws_size,
                              hipStream_t stream) {
  (void)in_sizes; (void)n_in; (void)out_size; (void)ws_size;

  const float* q  = (const float*)d_in[0];
  const float* k  = (const float*)d_in[1];
  const float* v  = (const float*)d_in[2];
  const float* Wq = (const float*)d_in[3];
  const float* bq = (const float*)d_in[4];
  const float* Wk = (const float*)d_in[5];
  const float* bk = (const float*)d_in[6];
  const float* Wv = (const float*)d_in[7];
  const float* bv = (const float*)d_in[8];

  float* ctx  = (float*)d_out;                          // [B][T][H] f32
  float* attn = ctx + (size_t)NBATCH * SEQ * HID;       // [B][S][T] f32 (also
                                                        //  scoresT scratch)

  // workspace: qbf(16MB) kbf(16MB) vT(16MB) attnT(32MB) = 80MB bf16 staging
  unsigned short* qbf   = (unsigned short*)d_ws;
  unsigned short* kbf   = qbf + (size_t)NBATCH * SEQ * HID;
  unsigned short* vT    = kbf + (size_t)NBATCH * SEQ * HID;
  unsigned short* attnT = vT  + (size_t)NBATCH * SEQ * HID;

  const dim3 blk(256);
  const float scale = 0.03125f;   // 1/sqrt(1024)

  // --- projections: (B*S, H) x W^T + b ---------------------------------
  const dim3 gProj(HID / 64, (NBATCH * SEQ) / 128, 1);
  gemm_bt<float, true, false, true, false><<<gProj, blk, 0, stream>>>(
      q, Wq, bq, qbf, HID, HID, HID, HID, 1.0f, 0, 0, 0);
  gemm_bt<float, true, false, true, false><<<gProj, blk, 0, stream>>>(
      k, Wk, bk, kbf, HID, HID, HID, HID, 1.0f, 0, 0, 0);
  gemm_bt<float, true, false, true, true><<<gProj, blk, 0, stream>>>(
      v, Wv, bv, vT, HID, HID, HID, 0, 1.0f, 0, 0, 0);   // -> vT[b][h][s]

  // --- scoresT[b][t][s] = scale * q_t . k_s  (+ causal -1e7) ------------
  const dim3 gScore(SEQ / 64, SEQ / 128, NBATCH);
  gemm_bt<unsigned short, false, true, false, false><<<gScore, blk, 0, stream>>>(
      qbf, kbf, nullptr, attn, HID, HID, HID, SEQ, scale,
      (long)SEQ * HID, (long)SEQ * HID, (long)SEQ * SEQ);

  // --- softmax over key dim (rows of scoresT) -> attnT bf16 -------------
  softmax_rows<<<dim3((NBATCH * SEQ) / 8), blk, 0, stream>>>(attn, attnT, SEQ);

  // --- emit reference-order attn f32 [b][s][t] --------------------------
  transpose_attn<<<dim3(SEQ / 32, SEQ / 32, NBATCH), dim3(32, 8), 0, stream>>>(
      attnT, attn, SEQ);

  // --- context[b][t][h] = attnT @ v  ------------------------------------
  const dim3 gCtx(HID / 64, SEQ / 128, NBATCH);
  gemm_bt<unsigned short, false, false, false, false><<<gCtx, blk, 0, stream>>>(
      attnT, vT, nullptr, ctx, SEQ, SEQ, SEQ, HID, 1.0f,
      (long)SEQ * SEQ, (long)SEQ * HID, (long)SEQ * HID);
}
